// Encoder_10101763080427
// MI455X (gfx1250) — compile-verified, matrix operand order checked by reading, not run on hardware
//
#include <hip/hip_runtime.h>
#include <cstdint>
#include <cstddef>

// ---------------------------------------------------------------------------
// Types for CDNA5 WMMA (wave32): bf16 A/B fragments (v16bf), f32 accum (v8f).
// ---------------------------------------------------------------------------
typedef __bf16 bf16;
typedef __attribute__((ext_vector_type(8)))  __bf16 v8bf;
typedef __attribute__((ext_vector_type(16))) __bf16 v16bf;
typedef __attribute__((ext_vector_type(8)))  float  v8f;
typedef __attribute__((ext_vector_type(4)))  int    v4i;

// Async-to-LDS builtin takes b128 payload pointers: AS1 (global) src, AS3 (LDS) dst.
typedef __attribute__((address_space(1))) v4i* as1_v4i_p;
typedef __attribute__((address_space(3))) v4i* as3_v4i_p;

// gfx1250 async global->LDS copies (ASYNCcnt path), guarded for portability.
#if defined(__HIP_DEVICE_COMPILE__) && \
    __has_builtin(__builtin_amdgcn_global_load_async_to_lds_b128) && \
    __has_builtin(__builtin_amdgcn_s_wait_asynccnt)
#define HAVE_ASYNC_LDS 1
#else
#define HAVE_ASYNC_LDS 0
#endif

#if defined(__HIP_DEVICE_COMPILE__) && __has_builtin(__builtin_amdgcn_update_dpp)
#define HAVE_DPP 1
#else
#define HAVE_DPP 0
#endif

__device__ __forceinline__ void async_copy16(bf16* lds_dst, const bf16* gsrc) {
#if HAVE_ASYNC_LDS
    __builtin_amdgcn_global_load_async_to_lds_b128(
        (as1_v4i_p)gsrc, (as3_v4i_p)lds_dst, /*offset=*/0, /*cpol=*/0);
#else
    *(v8bf*)lds_dst = *(const v8bf*)gsrc;
#endif
}

__device__ __forceinline__ void async_wait_all() {
#if HAVE_ASYNC_LDS
    __builtin_amdgcn_s_wait_asynccnt(0);
#endif
}

__device__ __forceinline__ v8f v8f_zero() {
    v8f z = {0.f, 0.f, 0.f, 0.f, 0.f, 0.f, 0.f, 0.f};
    return z;
}

__device__ __forceinline__ v8f wmma_bf16(v16bf a, v16bf b, v8f c) {
    return __builtin_amdgcn_wmma_f32_16x16x32_bf16(
        false, a, false, b, (short)0, c, false, false);
}

// 16x32 bf16 tile (row-major, 'stride' elems/row) -> A-matrix per-lane layout:
//   lanes 0-15 : row = lane,    K = {0..7, 16..23}
//   lanes 16-31: row = lane-16, K = {8..15, 24..31}
// Also used for B fragments when B is staged N-major.
__device__ __forceinline__ v16bf load_frag16x32(const bf16* base, int stride) {
    const int lane = threadIdx.x & 31;
    const int row  = lane & 15;
    const int half = (lane >> 4) << 3;
    const bf16* p  = base + row * stride + half;
    v8bf lo = *(const v8bf*)(p);
    v8bf hi = *(const v8bf*)(p + 16);
    return __builtin_shufflevector(lo, hi,
        0, 1, 2, 3, 4, 5, 6, 7, 8, 9, 10, 11, 12, 13, 14, 15);
}

// VALU-only max-reduction across a 16-lane DPP row (no LDS round trips).
#if HAVE_DPP
template <int CTRL>
__device__ __forceinline__ float dpp_mov_f(float v) {
    return __int_as_float(
        __builtin_amdgcn_update_dpp(0, __float_as_int(v), CTRL, 0xF, 0xF, true));
}
#endif

__device__ __forceinline__ float row16_max(float x) {
#if HAVE_DPP
    x = fmaxf(x, dpp_mov_f<0xB1>(x));    // quad_perm [1,0,3,2]  (xor 1)
    x = fmaxf(x, dpp_mov_f<0x4E>(x));    // quad_perm [2,3,0,1]  (xor 2)
    x = fmaxf(x, dpp_mov_f<0x141>(x));   // row_half_mirror      (8-span)
    x = fmaxf(x, dpp_mov_f<0x140>(x));   // row_mirror           (16-span)
#else
    x = fmaxf(x, __shfl_xor(x, 1));
    x = fmaxf(x, __shfl_xor(x, 2));
    x = fmaxf(x, __shfl_xor(x, 4));
    x = fmaxf(x, __shfl_xor(x, 8));
#endif
    return x;
}

// ---------------------------------------------------------------------------
// Weight preprocessing (once per call): f32 -> bf16 with transpose so every
// GEMM B-operand is already N-major and LDS staging is a straight copy.
// ---------------------------------------------------------------------------
// in [R][Cc] f32 -> out [Cc][R] bf16
__global__ void transpose_cvt_kernel(const float* __restrict__ in,
                                     bf16* __restrict__ out, int R, int Cc) {
    int i = blockIdx.x * 256 + threadIdx.x;
    if (i >= R * Cc) return;
    int r = i / Cc, c = i % Cc;
    out[(size_t)c * R + r] = (bf16)in[i];
}

// wq/wk/wv (H, C, D) f32 -> [H*D][C] bf16 (transposed head-concat weight)
__global__ void repack_hcd_t_kernel(const float* __restrict__ w,
                                    bf16* __restrict__ o,
                                    int H, int C, int D) {
    int i = blockIdx.x * 256 + threadIdx.x;
    if (i >= H * C * D) return;
    int d = i % D;
    int c = (i / D) % C;
    int h = i / (D * C);
    o[(size_t)(h * D + d) * C + c] = (bf16)w[i];
}

// ---------------------------------------------------------------------------
// LayerNorm over C=1024 -> bf16. One row per 256-thread block.
// ---------------------------------------------------------------------------
__global__ __launch_bounds__(256)
void ln_bf16_kernel(const float* __restrict__ x, const float* __restrict__ g,
                    const float* __restrict__ b, bf16* __restrict__ out, int C) {
    __shared__ float red[256];
    const int row = blockIdx.x;
    const int tid = threadIdx.x;
    const float* xr = x + (size_t)row * C;
    float v[4];
    float s = 0.f;
#pragma unroll
    for (int i = 0; i < 4; i++) { v[i] = xr[tid + i * 256]; s += v[i]; }
    red[tid] = s;
    __syncthreads();
    for (int o = 128; o > 0; o >>= 1) {
        if (tid < o) red[tid] += red[tid + o];
        __syncthreads();
    }
    const float mu = red[0] / (float)C;
    __syncthreads();
    float s2 = 0.f;
#pragma unroll
    for (int i = 0; i < 4; i++) { float d = v[i] - mu; s2 += d * d; }
    red[tid] = s2;
    __syncthreads();
    for (int o = 128; o > 0; o >>= 1) {
        if (tid < o) red[tid] += red[tid + o];
        __syncthreads();
    }
    const float rstd = rsqrtf(red[0] / (float)C + 1e-5f);
#pragma unroll
    for (int i = 0; i < 4; i++) {
        int c = tid + i * 256;
        out[(size_t)row * C + c] = (bf16)((v[i] - mu) * rstd * g[c] + b[c]);
    }
}

// ---------------------------------------------------------------------------
// Tiled WMMA GEMM: out[M,N] = A[M,K] * BT[N,K]^T  (+bias)(+relu)(+resid)
// Block 128x128, 8 waves (4Mx2N), wave tile 32x64, K-step 32.
// Double-buffered LDS; both operands staged with 16B async copies.
// OUT_TRANS stores out[n*M + m] (used to produce V^T for attention).
// ---------------------------------------------------------------------------
template <bool RELU, bool RESID, bool OUT_BF16, bool OUT_TRANS>
__global__ __launch_bounds__(256)
void wmma_gemm_kernel(const bf16* __restrict__ A, const bf16* __restrict__ BT,
                      const float* __restrict__ bias,
                      const float* __restrict__ resid,
                      void* __restrict__ out, int M, int N, int K) {
    constexpr int BM = 128, BN = 128, BK = 32;
    constexpr int LDT = BK + 8;   // 40: pad, keeps 16B alignment
    __shared__ __align__(16) bf16 As[2][BM * LDT];
    __shared__ __align__(16) bf16 Bs[2][BN * LDT];   // [n][k]

    const int tid  = threadIdx.x;
    const int lane = tid & 31;
    const int wid  = tid >> 5;
    const int wm   = wid & 3;
    const int wn   = wid >> 2;
    const int mb   = blockIdx.x * BM;
    const int nb   = blockIdx.y * BN;

    const int srow = tid >> 1;            // 0..127
    const int skc  = (tid & 1) * 16;      // 0 or 16

    auto stage = [&](int buf, int kb) {
        const bf16* ga = A  + (size_t)(mb + srow) * K + kb + skc;
        const bf16* gb = BT + (size_t)(nb + srow) * K + kb + skc;
        bf16* la = &As[buf][srow * LDT + skc];
        bf16* lb = &Bs[buf][srow * LDT + skc];
        async_copy16(la,     ga);
        async_copy16(la + 8, ga + 8);
        async_copy16(lb,     gb);
        async_copy16(lb + 8, gb + 8);
    };

    v8f acc[2][4];
#pragma unroll
    for (int i = 0; i < 2; i++)
#pragma unroll
        for (int j = 0; j < 4; j++) acc[i][j] = v8f_zero();

    stage(0, 0);
    async_wait_all();
    __syncthreads();

    int buf = 0;
    for (int kb = 0; kb < K; kb += BK, buf ^= 1) {
        const bool has_next = (kb + BK) < K;
        if (has_next) stage(buf ^ 1, kb + BK);

        v16bf afrag[2], bfrag[4];
#pragma unroll
        for (int i = 0; i < 2; i++)
            afrag[i] = load_frag16x32(&As[buf][(wm * 32 + i * 16) * LDT], LDT);
#pragma unroll
        for (int j = 0; j < 4; j++)
            bfrag[j] = load_frag16x32(&Bs[buf][(wn * 64 + j * 16) * LDT], LDT);
#pragma unroll
        for (int i = 0; i < 2; i++)
#pragma unroll
            for (int j = 0; j < 4; j++)
                acc[i][j] = wmma_bf16(afrag[i], bfrag[j], acc[i][j]);

        if (has_next) {
            async_wait_all();
            __syncthreads();
        }
    }

    // Epilogue. C-layout: VGPR r -> row = r + 8*(lane>>4), col = lane&15.
    const int rbase = (lane >> 4) << 3;
    const int nl    = lane & 15;
#pragma unroll
    for (int i = 0; i < 2; i++) {
#pragma unroll
        for (int j = 0; j < 4; j++) {
            const int col = nb + wn * 64 + j * 16 + nl;
            const float bv = bias ? bias[col] : 0.f;
#pragma unroll
            for (int r = 0; r < 8; r++) {
                const int row = mb + wm * 32 + i * 16 + r + rbase;
                float v = acc[i][j][r] + bv;
                if (RELU) v = v > 0.f ? v : 0.f;
                if (RESID) v += resid[(size_t)row * N + col];
                const size_t idx = OUT_TRANS ? ((size_t)col * M + row)
                                             : ((size_t)row * N + col);
                if (OUT_BF16) ((bf16*)out)[idx] = (bf16)v;
                else          ((float*)out)[idx] = v;
            }
        }
    }
}

// ---------------------------------------------------------------------------
// Flash attention (online softmax), bf16 WMMA, wave32.
// Q,K: [B*T, H*D] row-major; Vt: [H*D, B*T] (pre-transposed by the V GEMM);
// O: [B*T, H*D]. grid = (B*H, T/128); 8 waves, wave owns 16 queries.
// Keys streamed in 32-chunks; double-buffered async LDS staging.
// Row-sum of P comes from an extra WMMA against an all-ones B fragment, so
// the only cross-lane op is a 4-step DPP max.
// ---------------------------------------------------------------------------
__global__ __launch_bounds__(256)
void flash_attn_kernel(const bf16* __restrict__ Q, const bf16* __restrict__ K,
                       const bf16* __restrict__ Vt, bf16* __restrict__ O,
                       int Bn, int T, int Hh, int D, int Mtok, float scale) {
    constexpr int LKV = 64 + 8;   // K chunk: [s][d]
    constexpr int LVT = 32 + 8;   // V^T chunk: [d][s]
    constexpr int LP  = 32 + 8;   // per-wave P tile: [q][s]
    __shared__ __align__(16) bf16 ldsK[2][32 * LKV];
    __shared__ __align__(16) bf16 ldsVt[2][64 * LVT];
    __shared__ __align__(16) bf16 ldsP[8 * 16 * LP];

    const int tid  = threadIdx.x;
    const int lane = tid & 31;
    const int wid  = tid >> 5;
    const int bh   = blockIdx.x;
    const int b    = bh / Hh;
    const int h    = bh % Hh;
    const int q0   = blockIdx.y * 128 + wid * 16;
    const int HD   = Hh * D;

    // Resident Q fragments (16 q x 64 d), with softmax scale folded in
    // (scale = 2^-5, exact in bf16).
    v16bf qf[2];
    {
        const bf16* Qbase = Q + (size_t)(b * T + q0) * HD + h * D;
        const int row  = lane & 15;
        const int half = (lane >> 4) << 3;
        const bf16* p  = Qbase + (size_t)row * HD + half;
        v8bf a0 = *(const v8bf*)(p);
        v8bf a1 = *(const v8bf*)(p + 16);
        v8bf a2 = *(const v8bf*)(p + 32);
        v8bf a3 = *(const v8bf*)(p + 48);
        qf[0] = __builtin_shufflevector(a0, a1, 0,1,2,3,4,5,6,7,8,9,10,11,12,13,14,15);
        qf[1] = __builtin_shufflevector(a2, a3, 0,1,2,3,4,5,6,7,8,9,10,11,12,13,14,15);
#pragma unroll
        for (int j = 0; j < 2; j++)
#pragma unroll
            for (int i = 0; i < 16; i++)
                qf[j][i] = (bf16)((float)qf[j][i] * scale);
    }

    v16bf ones;
#pragma unroll
    for (int i = 0; i < 16; i++) ones[i] = (bf16)1.0f;

    v8f o_acc[4];
#pragma unroll
    for (int j = 0; j < 4; j++) o_acc[j] = v8f_zero();
    v8f o_l = v8f_zero();                 // running row-sums via ones-WMMA
    float m_r[8];
#pragma unroll
    for (int r = 0; r < 8; r++) m_r[r] = -1e30f;

    const int rbase = (lane >> 4) << 3;
    const int nl    = lane & 15;

    const int ks = tid >> 3, kd0 = (tid & 7) * 8;   // K staging map
    const int vd = tid >> 2, vs0 = (tid & 3) * 8;   // Vt staging map

    auto stage = [&](int buf, int s0c) {
        async_copy16(&ldsK[buf][ks * LKV + kd0],
                     K + (size_t)(b * T + s0c + ks) * HD + h * D + kd0);
        async_copy16(&ldsVt[buf][vd * LVT + vs0],
                     Vt + (size_t)(h * D + vd) * Mtok + b * T + s0c + vs0);
    };

    stage(0, 0);
    async_wait_all();
    __syncthreads();

    int buf = 0;
    for (int s0 = 0; s0 < T; s0 += 32, buf ^= 1) {
        const bool has_next = (s0 + 32) < T;
        if (has_next) stage(buf ^ 1, s0 + 32);

        // S = (Q*scale) x K^T : two 16x16 tiles, K-dim 64.
        v8f sa[2];
#pragma unroll
        for (int nt = 0; nt < 2; nt++) {
            sa[nt] = v8f_zero();
            v16bf b0 = load_frag16x32(&ldsK[buf][(nt * 16) * LKV + 0],  LKV);
            v16bf b1 = load_frag16x32(&ldsK[buf][(nt * 16) * LKV + 32], LKV);
            sa[nt] = wmma_bf16(qf[0], b0, sa[nt]);
            sa[nt] = wmma_bf16(qf[1], b1, sa[nt]);
        }

        // Online softmax: DPP row-max, exp, rescale accumulators.
        float pr[2][8];
#pragma unroll
        for (int r = 0; r < 8; r++) {
            const float s0v = sa[0][r];
            const float s1v = sa[1][r];
            const float mx   = row16_max(fmaxf(s0v, s1v));
            const float mnew = fmaxf(m_r[r], mx);
            const float corr = __expf(m_r[r] - mnew);
            pr[0][r] = __expf(s0v - mnew);
            pr[1][r] = __expf(s1v - mnew);
            m_r[r] = mnew;
            o_l[r] *= corr;
#pragma unroll
            for (int j = 0; j < 4; j++) o_acc[j][r] *= corr;
        }

        // C-layout -> A-layout round trip of P via per-wave LDS tile.
        bf16* P = &ldsP[wid * 16 * LP];
#pragma unroll
        for (int nt = 0; nt < 2; nt++)
#pragma unroll
            for (int r = 0; r < 8; r++)
                P[(r + rbase) * LP + nt * 16 + nl] = (bf16)pr[nt][r];

        v16bf pf = load_frag16x32(P, LP);           // 16 q x 32 s
        o_l = wmma_bf16(pf, ones, o_l);             // row sums for free
#pragma unroll
        for (int j = 0; j < 4; j++) {
            v16bf vf = load_frag16x32(&ldsVt[buf][(j * 16) * LVT], LVT);
            o_acc[j] = wmma_bf16(pf, vf, o_acc[j]);
        }

        if (has_next) {
            async_wait_all();
            __syncthreads();
        }
    }

    bf16* Obase = O + (size_t)(b * T + q0) * HD + h * D;
#pragma unroll
    for (int j = 0; j < 4; j++)
#pragma unroll
        for (int r = 0; r < 8; r++) {
            float v = o_acc[j][r] / o_l[r];
            Obase[(size_t)(r + rbase) * HD + j * 16 + nl] = (bf16)v;
        }
}

// ---------------------------------------------------------------------------
// Host-side orchestration
// ---------------------------------------------------------------------------
extern "C" void kernel_launch(void* const* d_in, const int* in_sizes, int n_in,
                              void* d_out, int out_size, void* d_ws, size_t ws_size,
                              hipStream_t stream) {
    (void)in_sizes; (void)n_in; (void)out_size; (void)ws_size;

    constexpr int Bn = 4, T = 2048, C = 1024, H = 16, D = 64;
    constexpr int M  = Bn * T;        // 8192
    constexpr int C4 = 4 * C;         // 4096

    const float* x      = (const float*)d_in[0];
    const float* wq     = (const float*)d_in[1];
    const float* wk     = (const float*)d_in[2];
    const float* wv     = (const float*)d_in[3];
    const float* w_proj = (const float*)d_in[4];
    const float* b_proj = (const float*)d_in[5];
    const float* w1     = (const float*)d_in[6];
    const float* b1     = (const float*)d_in[7];
    const float* w2     = (const float*)d_in[8];
    const float* b2     = (const float*)d_in[9];
    const float* g1     = (const float*)d_in[10];
    const float* be1    = (const float*)d_in[11];
    const float* g2     = (const float*)d_in[12];
    const float* be2    = (const float*)d_in[13];

    char* ws = (char*)d_ws;
    size_t off = 0;
    auto carve = [&](size_t bytes) -> void* {
        void* p = ws + off;
        off += (bytes + 255) & ~(size_t)255;
        return p;
    };

    bf16*  WqT  = (bf16*)carve((size_t)C * C * 2);    // [H*D][C]
    bf16*  WkT  = (bf16*)carve((size_t)C * C * 2);
    bf16*  WvT  = (bf16*)carve((size_t)C * C * 2);
    bf16*  WpT  = (bf16*)carve((size_t)C * C * 2);    // [C][C]
    bf16*  W1T  = (bf16*)carve((size_t)C * C4 * 2);   // [4C][C]
    bf16*  W2T  = (bf16*)carve((size_t)C4 * C * 2);   // [C][4C]
    bf16*  hbf  = (bf16*)carve((size_t)M * C * 2);
    bf16*  qbf  = (bf16*)carve((size_t)M * C * 2);
    bf16*  kbf  = (bf16*)carve((size_t)M * C * 2);
    bf16*  vbfT = (bf16*)carve((size_t)M * C * 2);    // [H*D][B*T]
    bf16*  att  = (bf16*)carve((size_t)M * C * 2);
    float* x1   = (float*)carve((size_t)M * C * 4);
    bf16*  h2bf = (bf16*)carve((size_t)M * C * 2);
    bf16*  ff1  = (bf16*)carve((size_t)M * C4 * 2);

    // --- weight conversion / transpose (once per call) ---
    {
        dim3 g((H * C * D + 255) / 256);
        repack_hcd_t_kernel<<<g, 256, 0, stream>>>(wq, WqT, H, C, D);
        repack_hcd_t_kernel<<<g, 256, 0, stream>>>(wk, WkT, H, C, D);
        repack_hcd_t_kernel<<<g, 256, 0, stream>>>(wv, WvT, H, C, D);
    }
    transpose_cvt_kernel<<<dim3((C * C + 255) / 256), 256, 0, stream>>>(w_proj, WpT, C, C);
    transpose_cvt_kernel<<<dim3((C * C4 + 255) / 256), 256, 0, stream>>>(w1, W1T, C, C4);
    transpose_cvt_kernel<<<dim3((C4 * C + 255) / 256), 256, 0, stream>>>(w2, W2T, C4, C);

    // --- h = ln1(x) ---
    ln_bf16_kernel<<<dim3(M), 256, 0, stream>>>(x, g1, be1, hbf, C);

    // --- q,k = h @ W{q,k}; v stored transposed for attention ---
    {
        dim3 g(M / 128, C / 128);
        wmma_gemm_kernel<false, false, true, false><<<g, 256, 0, stream>>>(
            hbf, WqT, nullptr, nullptr, qbf, M, C, C);
        wmma_gemm_kernel<false, false, true, false><<<g, 256, 0, stream>>>(
            hbf, WkT, nullptr, nullptr, kbf, M, C, C);
        wmma_gemm_kernel<false, false, true, true><<<g, 256, 0, stream>>>(
            hbf, WvT, nullptr, nullptr, vbfT, M, C, C);
    }

    // --- attention (scale = C^-0.5 = 2^-5 per reference) ---
    flash_attn_kernel<<<dim3(Bn * H, T / 128), 256, 0, stream>>>(
        qbf, kbf, vbfT, att, Bn, T, H, D, M, 0.03125f);

    // --- x1 = x + att @ Wp + b_proj ---
    wmma_gemm_kernel<false, true, false, false><<<dim3(M / 128, C / 128), 256, 0, stream>>>(
        att, WpT, b_proj, x, x1, M, C, C);

    // --- h2 = ln2(x1) ---
    ln_bf16_kernel<<<dim3(M), 256, 0, stream>>>(x1, g2, be2, h2bf, C);

    // --- ff1 = relu(h2 @ W1 + b1) ---
    wmma_gemm_kernel<true, false, true, false><<<dim3(M / 128, C4 / 128), 256, 0, stream>>>(
        h2bf, W1T, b1, nullptr, ff1, M, C4, C);

    // --- out = x1 + ff1 @ W2 + b2 ---
    wmma_gemm_kernel<false, true, false, false><<<dim3(M / 128, C / 128), 256, 0, stream>>>(
        ff1, W2T, b2, x1, (float*)d_out, M, C, C4);
}
